// Mamba5Block_2731599200637
// MI455X (gfx1250) — matrix operand order repaired and checked by reading, not compiled, not run on hardware
//
#include <hip/hip_runtime.h>
#include <hip/hip_bf16.h>
#include <math.h>

// ---------------- problem constants ----------------
#define Bc 2
#define Tc 64
#define Nc 256
#define DMc 128
#define Hc 8
#define HDc 16
#define DFFc 512
#define Ec 4
#define WINc 4
#define DSTATEc 16
#define DCONVc 4
#define DIc 256
#define DTRANKc 8
#define EPSc 1e-6f
#define BNc (Bc * Nc)          // 512 sequences (mamba)
#define ROWSc (BNc * Tc)       // 32768 rows in both stages

typedef __attribute__((ext_vector_type(16))) __bf16 v16bf;
typedef __attribute__((ext_vector_type(8)))  __bf16 v8bf;
typedef __attribute__((ext_vector_type(8)))  float  v8f;

static __device__ __forceinline__ __bf16 tobf(float x) { return (__bf16)x; }

static __device__ __forceinline__ int map_rev(int m, int Tlen, int rev) {
  if (!rev) return m;
  int s = m / Tlen, t = m - s * Tlen;
  return s * Tlen + (Tlen - 1 - t);
}

// load 8 consecutive floats (two b128) -> bf16 into frag[base..base+7], scaled
static __device__ __forceinline__ void ld8bf(const float* __restrict__ p,
                                             v16bf& f, int base, float msk) {
  const float4 a = *(const float4*)(p);
  const float4 b = *(const float4*)(p + 4);
  f[base + 0] = tobf(a.x * msk); f[base + 1] = tobf(a.y * msk);
  f[base + 2] = tobf(a.z * msk); f[base + 3] = tobf(a.w * msk);
  f[base + 4] = tobf(b.x * msk); f[base + 5] = tobf(b.y * msk);
  f[base + 6] = tobf(b.z * msk); f[base + 7] = tobf(b.w * msk);
}

// ---------------- generic bf16 WMMA GEMM: C = A(MxK) * W(NxK)^T [+Cin] ------
// one wave computes a 64x16 tile (4 M-subtiles share each B fragment).
// A may be split at ksplit (multiple of 32) between A0/A1; optional per-seq
// time reversal of A rows (revA) or of store rows (revC); N guarded by Nw.
__global__ __launch_bounds__(32)
void k_gemm(const float* __restrict__ A0, const float* __restrict__ A1, int ksplit,
            int lda0, int lda1,
            const float* __restrict__ W, int ldw, int Nw,
            const float* __restrict__ Cin, float* __restrict__ Cout, int ldc,
            int K, int Tlen, int revA, int revC)
{
  const int n0 = blockIdx.x * 16;
  const int m0 = blockIdx.y * 64;
  const int lane = threadIdx.x;
  const int r  = lane & 15;
  const int hi = lane >> 4;

  int arow[4];
#pragma unroll
  for (int tt = 0; tt < 4; ++tt)
    arow[tt] = map_rev(m0 + tt * 16 + r, Tlen, revA);

  const int  ncol_raw = n0 + r;
  const float bmask = (ncol_raw < Nw) ? 1.f : 0.f;
  const int  ncol   = (ncol_raw < Nw) ? ncol_raw : (Nw - 1);
  const float* __restrict__ Wrow = W + (size_t)ncol * ldw;

  v8f acc[4] = {{}, {}, {}, {}};
  for (int kk = 0; kk < K; kk += 32) {
    // B fragment: 8 floats at k = kk + hi*8 and 8 at k = kk + 16 + hi*8
    v16bf b;
    {
      const float* bp = Wrow + kk + hi * 8;
      ld8bf(bp,      b, 0, bmask);
      ld8bf(bp + 16, b, 8, bmask);
    }
    const int g0 = kk + hi * 8;        // chunk of 8, never crosses ksplit
    const int g1 = kk + 16 + hi * 8;
#pragma unroll
    for (int tt = 0; tt < 4; ++tt) {
      v16bf a;
      const float* p0 = (g0 < ksplit) ? (A0 + (size_t)arow[tt] * lda0 + g0)
                                      : (A1 + (size_t)arow[tt] * lda1 + (g0 - ksplit));
      const float* p1 = (g1 < ksplit) ? (A0 + (size_t)arow[tt] * lda0 + g1)
                                      : (A1 + (size_t)arow[tt] * lda1 + (g1 - ksplit));
      ld8bf(p0, a, 0, 1.f);
      ld8bf(p1, a, 8, 1.f);
      acc[tt] = __builtin_amdgcn_wmma_f32_16x16x32_bf16(false, a, false, b,
                                                        (short)0, acc[tt],
                                                        false, false);
    }
  }
  const int col = n0 + r;
  if (col < Nw) {
#pragma unroll
    for (int tt = 0; tt < 4; ++tt) {
#pragma unroll
      for (int i = 0; i < 8; ++i) {
        const int m = m0 + tt * 16 + hi * 8 + i;
        const int crow = map_rev(m, Tlen, revC);
        float v = acc[tt][i];
        if (Cin) v += Cin[(size_t)crow * ldc + col];
        Cout[(size_t)crow * ldc + col] = v;
      }
    }
  }
}

// ---------------- transpose (B,T,N,D)->(B*N,T,D) + RMSNorm ----------------
__global__ __launch_bounds__(128)
void k_t_norm(const float* __restrict__ x, const float* __restrict__ w,
              float* __restrict__ xr, float* __restrict__ xn)
{
  const int row = blockIdx.x;            // (b*N+n)*T + t
  const int d = threadIdx.x;
  const int t = row % Tc;
  const int bn = row / Tc;
  const int n = bn % Nc, b = bn / Nc;
  const float v = x[(((size_t)b * Tc + t) * Nc + n) * DMc + d];
  __shared__ float red[4];
  float s = v * v;
  for (int o = 16; o > 0; o >>= 1) s += __shfl_down(s, o, 32);
  if ((threadIdx.x & 31) == 0) red[threadIdx.x >> 5] = s;
  __syncthreads();
  const float tot = red[0] + red[1] + red[2] + red[3];
  const float rs = rsqrtf(tot / DMc + EPSc);
  xr[(size_t)row * DMc + d] = v;
  xn[(size_t)row * DMc + d] = v * rs * w[d];
}

// ---------------- transpose (B*N,T,D)->(B*T,N,D) + RMSNorm ----------------
__global__ __launch_bounds__(128)
void k_t2_norm(const float* __restrict__ xr, const float* __restrict__ w,
               float* __restrict__ xa, float* __restrict__ xan)
{
  const int m = blockIdx.x;              // (b*T+t)*N + n
  const int d = threadIdx.x;
  const int n = m % Nc;
  const int bt = m / Nc;
  const int t = bt % Tc, b = bt / Tc;
  const float v = xr[(((size_t)b * Nc + n) * Tc + t) * DMc + d];
  __shared__ float red[4];
  float s = v * v;
  for (int o = 16; o > 0; o >>= 1) s += __shfl_down(s, o, 32);
  if ((threadIdx.x & 31) == 0) red[threadIdx.x >> 5] = s;
  __syncthreads();
  const float tot = red[0] + red[1] + red[2] + red[3];
  const float rs = rsqrtf(tot / DMc + EPSc);
  xa[(size_t)m * DMc + d] = v;
  xan[(size_t)m * DMc + d] = v * rs * w[d];
}

// ---------------- plain RMSNorm ----------------
__global__ __launch_bounds__(128)
void k_rmsnorm(const float* __restrict__ in, const float* __restrict__ w,
               float* __restrict__ out)
{
  const int row = blockIdx.x;
  const int d = threadIdx.x;
  const float v = in[(size_t)row * DMc + d];
  __shared__ float red[4];
  float s = v * v;
  for (int o = 16; o > 0; o >>= 1) s += __shfl_down(s, o, 32);
  if ((threadIdx.x & 31) == 0) red[threadIdx.x >> 5] = s;
  __syncthreads();
  const float tot = red[0] + red[1] + red[2] + red[3];
  const float rs = rsqrtf(tot / DMc + EPSc);
  out[(size_t)row * DMc + d] = v * rs * w[d];
}

// ---------------- causal depthwise conv (width 4) + bias + SiLU ------------
__global__ void k_conv_silu(const float* __restrict__ xz,
                            const float* __restrict__ cw,
                            const float* __restrict__ cb,
                            float* __restrict__ xc, int dir)
{
  const size_t i = (size_t)blockIdx.x * blockDim.x + threadIdx.x;
  if (i >= (size_t)BNc * Tc * DIc) return;
  const int d = (int)(i % DIc);
  const size_t st = i / DIc;
  const int t = (int)(st % Tc);
  const size_t seq = st / Tc;
  const float4 w = *(const float4*)(cw + ((size_t)dir * DIc + d) * DCONVc);
  float acc = cb[dir * DIc + d];
  const float* base = xz + (seq * Tc) * (size_t)(2 * DIc) + d;
  if (t >= 3) acc += base[(size_t)(t - 3) * (2 * DIc)] * w.x;
  if (t >= 2) acc += base[(size_t)(t - 2) * (2 * DIc)] * w.y;
  if (t >= 1) acc += base[(size_t)(t - 1) * (2 * DIc)] * w.z;
  acc += base[(size_t)t * (2 * DIc)] * w.w;
  xc[i] = acc / (1.f + expf(-acc));
}

// -------- fused dt + selective scan + D skip + SiLU(z) gating --------------
// one block per sequence (256 threads = one channel each); xdbl row staged
// in LDS once per timestep; writes gated output into xz[..., 0:256].
__global__ __launch_bounds__(256)
void k_scan(const float* __restrict__ xz, const float* __restrict__ xc,
            const float* __restrict__ xdbl,
            const float* __restrict__ dt_w, const float* __restrict__ dt_b,
            const float* __restrict__ a_log, const float* __restrict__ Dp,
            float* __restrict__ gout, int dir)
{
  const int seq = blockIdx.x;
  const int d = threadIdx.x;            // 0..255
  __shared__ float srow[40];

  float A[DSTATEc];
#pragma unroll
  for (int s = 0; s < DSTATEc; ++s)
    A[s] = -expf(a_log[((size_t)dir * DIc + d) * DSTATEc + s]);
  float dtw[DTRANKc];
#pragma unroll
  for (int r = 0; r < DTRANKc; ++r)
    dtw[r] = dt_w[((size_t)dir * DIc + d) * DTRANKc + r];
  const float dtb = dt_b[dir * DIc + d];
  const float Dd = Dp[dir * DIc + d];
  float h[DSTATEc];
#pragma unroll
  for (int s = 0; s < DSTATEc; ++s) h[s] = 0.f;

  for (int t = 0; t < Tc; ++t) {
    __syncthreads();
    if (d < 40) srow[d] = xdbl[(size_t)(seq * Tc + t) * 40 + d];
    __syncthreads();
    float dta = dtb;
#pragma unroll
    for (int r = 0; r < DTRANKc; ++r) dta += srow[r] * dtw[r];
    const float dt = (dta > 20.f) ? dta : log1pf(expf(dta));
    const float xt = xc[((size_t)seq * Tc + t) * DIc + d];
    const float dtx = dt * xt;
    float y = 0.f;
#pragma unroll
    for (int s = 0; s < DSTATEc; ++s) {
      const float dA = expf(dt * A[s]);
      h[s] = dA * h[s] + dtx * srow[8 + s];
      y += h[s] * srow[24 + s];
    }
    y += Dd * xt;
    const float z = xz[((size_t)seq * Tc + t) * (2 * DIc) + DIc + d];
    const float sz = z / (1.f + expf(-z));
    gout[((size_t)seq * Tc + t) * (2 * DIc) + d] = y * sz;
  }
}

// ---------------- banded local attention (window +-4), one thread/(bt,n,h) --
__global__ void k_attn(const float* __restrict__ q, const float* __restrict__ k,
                       const float* __restrict__ v, float* __restrict__ o)
{
  const int idx = blockIdx.x * blockDim.x + threadIdx.x;
  if (idx >= Bc * Tc * Nc * Hc) return;
  const int h = idx % Hc;
  const int n = (idx / Hc) % Nc;
  const int bt = idx / (Hc * Nc);
  const float4* qp = (const float4*)(q + ((size_t)bt * Nc + n) * DMc + h * HDc);
  float4 q0 = qp[0], q1 = qp[1], q2 = qp[2], q3 = qp[3];
  const int j0 = (n - WINc > 0) ? n - WINc : 0;
  const int j1 = (n + WINc < Nc - 1) ? n + WINc : Nc - 1;
  float sc[2 * WINc + 1];
  float mx = -1e30f;
  for (int j = j0; j <= j1; ++j) {
    const float4* kp = (const float4*)(k + ((size_t)bt * Nc + j) * DMc + h * HDc);
    const float4 k0 = kp[0], k1 = kp[1], k2 = kp[2], k3 = kp[3];
    float s = q0.x * k0.x + q0.y * k0.y + q0.z * k0.z + q0.w * k0.w
            + q1.x * k1.x + q1.y * k1.y + q1.z * k1.z + q1.w * k1.w
            + q2.x * k2.x + q2.y * k2.y + q2.z * k2.z + q2.w * k2.w
            + q3.x * k3.x + q3.y * k3.y + q3.z * k3.z + q3.w * k3.w;
    s *= 0.25f;                      // HD^-0.5 = 1/4
    sc[j - j0] = s;
    mx = fmaxf(mx, s);
  }
  float den = 0.f;
  for (int j = j0; j <= j1; ++j) { sc[j - j0] = expf(sc[j - j0] - mx); den += sc[j - j0]; }
  float4 o0 = {0,0,0,0}, o1 = {0,0,0,0}, o2 = {0,0,0,0}, o3 = {0,0,0,0};
  for (int j = j0; j <= j1; ++j) {
    const float p = sc[j - j0] / den;
    const float4* vp = (const float4*)(v + ((size_t)bt * Nc + j) * DMc + h * HDc);
    const float4 v0 = vp[0], v1 = vp[1], v2 = vp[2], v3 = vp[3];
    o0.x += p * v0.x; o0.y += p * v0.y; o0.z += p * v0.z; o0.w += p * v0.w;
    o1.x += p * v1.x; o1.y += p * v1.y; o1.z += p * v1.z; o1.w += p * v1.w;
    o2.x += p * v2.x; o2.y += p * v2.y; o2.z += p * v2.z; o2.w += p * v2.w;
    o3.x += p * v3.x; o3.y += p * v3.y; o3.z += p * v3.z; o3.w += p * v3.w;
  }
  float4* op = (float4*)(o + ((size_t)bt * Nc + n) * DMc + h * HDc);
  op[0] = o0; op[1] = o1; op[2] = o2; op[3] = o3;
}

// ---------------- router: softmax over E, top-2, renormalize ----------------
__global__ void k_router(const float* __restrict__ xfn, const float* __restrict__ rw,
                         float* __restrict__ wfull)
{
  const int tkn = blockIdx.x * blockDim.x + threadIdx.x;
  if (tkn >= ROWSc) return;
  const float* xp = xfn + (size_t)tkn * DMc;
  float lg[Ec];
#pragma unroll
  for (int e = 0; e < Ec; ++e) {
    const float* w = rw + (size_t)e * DMc;
    float s = 0.f;
    for (int d = 0; d < DMc; ++d) s += xp[d] * w[d];
    lg[e] = s;
  }
  float mx = lg[0];
#pragma unroll
  for (int e = 1; e < Ec; ++e) mx = fmaxf(mx, lg[e]);
  float den = 0.f, p[Ec];
#pragma unroll
  for (int e = 0; e < Ec; ++e) { p[e] = expf(lg[e] - mx); den += p[e]; }
#pragma unroll
  for (int e = 0; e < Ec; ++e) p[e] /= den;
  int i0 = 0;
#pragma unroll
  for (int e = 1; e < Ec; ++e) if (p[e] > p[i0]) i0 = e;
  int i1 = -1;
#pragma unroll
  for (int e = 0; e < Ec; ++e)
    if (e != i0 && (i1 < 0 || p[e] > p[i1])) i1 = e;
  const float s2 = p[i0] + p[i1];
  float w4[Ec];
#pragma unroll
  for (int e = 0; e < Ec; ++e) w4[e] = 0.f;
  w4[i0] = p[i0] / s2;
  w4[i1] = p[i1] / s2;
#pragma unroll
  for (int e = 0; e < Ec; ++e) wfull[(size_t)tkn * Ec + e] = w4[e];
}

// -------- fused MoE: 16-token tile/block, hidden staged in LDS as bf16 ------
// out = xa + sum_e wfull[:,e] * ( gelu(xfn @ W1[e]^T) @ W2[e]^T )
__global__ __launch_bounds__(256)
void k_moe(const float* __restrict__ xfn, const float* __restrict__ xa,
           const float* __restrict__ wfull,
           const float* __restrict__ w1, const float* __restrict__ w2,
           float* __restrict__ out)
{
  __shared__ __align__(16) __bf16 lh[16 * DFFc];   // 16 tok x 512 hid = 16KB
  const int m0 = blockIdx.x * 16;
  const int lane = threadIdx.x & 31;
  const int wid  = threadIdx.x >> 5;     // 8 waves
  const int r  = lane & 15;
  const int hi = lane >> 4;

  // preload shared A fragments (16 tokens x 128 dm), 4 k-steps
  v16bf afrag[4];
#pragma unroll
  for (int kk = 0; kk < 4; ++kk) {
    const float* ap = xfn + (size_t)(m0 + r) * DMc + kk * 32;
    ld8bf(ap + hi * 8,      afrag[kk], 0, 1.f);
    ld8bf(ap + 16 + hi * 8, afrag[kk], 8, 1.f);
  }

  v8f accOut = {};
  for (int e = 0; e < Ec; ++e) {
    // phase 1: hidden = gelu(x @ W1^T); each wave does 4 of the 32 n-tiles
#pragma unroll
    for (int tt = 0; tt < 4; ++tt) {
      const int n0 = (wid + tt * 8) * 16;
      v8f acc = {};
#pragma unroll
      for (int kk = 0; kk < 4; ++kk) {
        v16bf b;
        const float* bp = w1 + ((size_t)e * DFFc + n0 + r) * DMc + kk * 32 + hi * 8;
        ld8bf(bp,      b, 0, 1.f);
        ld8bf(bp + 16, b, 8, 1.f);
        acc = __builtin_amdgcn_wmma_f32_16x16x32_bf16(false, afrag[kk], false, b,
                                                      (short)0, acc, false, false);
      }
#pragma unroll
      for (int i = 0; i < 8; ++i) {
        const int m = hi * 8 + i;
        const float xv = acc[i];
        const float g = 0.5f * xv * (1.f + erff(xv * 0.70710678f));
        lh[m * DFFc + n0 + r] = tobf(g);
      }
    }
    __syncthreads();
    // phase 2: out-tile wid: (16x16) = hidden(16x512) @ W2[e] chunk
    {
      const int n0 = wid * 16;
      v8f acc = {};
      for (int kk = 0; kk < DFFc; kk += 32) {
        v16bf a, b;
        const v8bf a0 = *(const v8bf*)&lh[r * DFFc + kk + hi * 8];
        const v8bf a1 = *(const v8bf*)&lh[r * DFFc + kk + 16 + hi * 8];
#pragma unroll
        for (int i = 0; i < 8; ++i) { a[i] = a0[i]; a[8 + i] = a1[i]; }
        const float* bp = w2 + ((size_t)e * DMc + n0 + r) * DFFc + kk + hi * 8;
        ld8bf(bp,      b, 0, 1.f);
        ld8bf(bp + 16, b, 8, 1.f);
        acc = __builtin_amdgcn_wmma_f32_16x16x32_bf16(false, a, false, b,
                                                      (short)0, acc, false, false);
      }
#pragma unroll
      for (int i = 0; i < 8; ++i) {
        const int m = hi * 8 + i;
        const float we = wfull[(size_t)(m0 + m) * Ec + e];
        accOut[i] += we * acc[i];
      }
    }
    __syncthreads();
  }
  // residual + store directly to d_out
  {
    const int n0 = wid * 16;
#pragma unroll
    for (int i = 0; i < 8; ++i) {
      const int m = hi * 8 + i;
      const size_t ix = (size_t)(m0 + m) * DMc + n0 + r;
      out[ix] = xa[ix] + accOut[i];
    }
  }
}

// ---------------- host orchestration ----------------
extern "C" void kernel_launch(void* const* d_in, const int* in_sizes, int n_in,
                              void* d_out, int out_size, void* d_ws, size_t ws_size,
                              hipStream_t stream) {
  (void)in_sizes; (void)n_in; (void)out_size; (void)ws_size;
  const float* x        = (const float*)d_in[0];
  const float* tn_w     = (const float*)d_in[1];
  const float* m_in_w   = (const float*)d_in[2];
  const float* m_conv_w = (const float*)d_in[3];
  const float* m_conv_b = (const float*)d_in[4];
  const float* m_xproj_w= (const float*)d_in[5];
  const float* m_dt_w   = (const float*)d_in[6];
  const float* m_dt_b   = (const float*)d_in[7];
  const float* m_A_log  = (const float*)d_in[8];
  const float* m_D      = (const float*)d_in[9];
  const float* m_out_w  = (const float*)d_in[10];
  const float* bi_out_w = (const float*)d_in[11];
  const float* sn_w     = (const float*)d_in[12];
  const float* q_w      = (const float*)d_in[13];
  const float* k_w      = (const float*)d_in[14];
  const float* v_w      = (const float*)d_in[15];
  const float* ao_w     = (const float*)d_in[16];
  const float* fn_w     = (const float*)d_in[17];
  const float* router_w = (const float*)d_in[18];
  const float* exp_w1   = (const float*)d_in[19];
  const float* exp_w2   = (const float*)d_in[20];

  float* ws = (float*)d_ws;
  const size_t SZ = (size_t)ROWSc * DMc;         // 4,194,304 floats
  float* xr   = ws;                              // residual, seq-major
  float* xn   = ws + SZ;                         // normed
  float* xz   = ws + 2 * SZ;                     // (ROWS,512) = 4*SZ floats
  float* xc   = ws + 6 * SZ;                     // (ROWS,256) = 2*SZ
  float* xdbl = ws + 8 * SZ;                     // (ROWS,40)
  float* y0   = xdbl + (size_t)ROWSc * 40;
  float* y1   = y0 + SZ;
  // attention-stage reuse of mamba buffers
  float* xa    = xn;
  float* xan   = xz;
  float* qb    = xz + SZ;
  float* kb    = xz + 2 * SZ;
  float* vb    = xz + 3 * SZ;
  float* attno = xc;
  float* xfn   = xc + SZ;
  float* wfull = xdbl;
  float* outp  = (float*)d_out;

  // 1) (B,T,N,D) -> (B*N,T,D) + RMSNorm
  k_t_norm<<<ROWSc, DMc, 0, stream>>>(x, tn_w, xr, xn);

  // 2) bidirectional Mamba
  for (int dir = 0; dir < 2; ++dir) {
    const float* inw = m_in_w + (size_t)dir * 2 * DIc * DMc;
    // in-proj: (ROWS,128)x(512,128)^T -> xz ; dir1 reads time-reversed rows
    k_gemm<<<dim3(2 * DIc / 16, ROWSc / 64), 32, 0, stream>>>(
        xn, xn, DMc, DMc, DMc, inw, DMc, 2 * DIc, nullptr, xz, 2 * DIc,
        DMc, Tc, dir, 0);
    // causal conv + SiLU
    k_conv_silu<<<((size_t)BNc * Tc * DIc + 255) / 256, 256, 0, stream>>>(
        xz, m_conv_w, m_conv_b, xc, dir);
    // x-proj: (ROWS,256)x(40,256)^T -> xdbl (N padded 40->48)
    const float* xpw = m_xproj_w + (size_t)dir * 40 * DIc;
    k_gemm<<<dim3(48 / 16, ROWSc / 64), 32, 0, stream>>>(
        xc, xc, DIc, DIc, DIc, xpw, DIc, 40, nullptr, xdbl, 40,
        DIc, Tc, 0, 0);
    // fused dt + scan + gate, writes g into xz[...,0:256]
    k_scan<<<BNc, DIc, 0, stream>>>(
        xz, xc, xdbl, m_dt_w, m_dt_b, m_A_log, m_D, xz, dir);
    // out-proj: (ROWS,256)x(128,256)^T -> y{dir} ; dir1 stores un-reversed
    const float* ow = m_out_w + (size_t)dir * DMc * DIc;
    k_gemm<<<dim3(DMc / 16, ROWSc / 64), 32, 0, stream>>>(
        xz, xz, DIc, 2 * DIc, 2 * DIc, ow, DIc, DMc, nullptr,
        dir ? y1 : y0, DMc, DIc, Tc, 0, dir);
  }

  // 3) bi-proj over concat(y0,y1) + residual into xr
  k_gemm<<<dim3(DMc / 16, ROWSc / 64), 32, 0, stream>>>(
      y0, y1, DMc, DMc, DMc, bi_out_w, 2 * DMc, DMc, xr, xr, DMc,
      2 * DMc, Tc, 0, 0);

  // 4) (B*N,T,D) -> (B*T,N,D) + RMSNorm
  k_t2_norm<<<ROWSc, DMc, 0, stream>>>(xr, sn_w, xa, xan);

  // 5) Q,K,V projections
  k_gemm<<<dim3(DMc / 16, ROWSc / 64), 32, 0, stream>>>(
      xan, xan, DMc, DMc, DMc, q_w, DMc, DMc, nullptr, qb, DMc, DMc, Tc, 0, 0);
  k_gemm<<<dim3(DMc / 16, ROWSc / 64), 32, 0, stream>>>(
      xan, xan, DMc, DMc, DMc, k_w, DMc, DMc, nullptr, kb, DMc, DMc, Tc, 0, 0);
  k_gemm<<<dim3(DMc / 16, ROWSc / 64), 32, 0, stream>>>(
      xan, xan, DMc, DMc, DMc, v_w, DMc, DMc, nullptr, vb, DMc, DMc, Tc, 0, 0);

  // 6) banded attention
  k_attn<<<(Bc * Tc * Nc * Hc + 255) / 256, 256, 0, stream>>>(qb, kb, vb, attno);

  // 7) AO projection + residual into xa
  k_gemm<<<dim3(DMc / 16, ROWSc / 64), 32, 0, stream>>>(
      attno, attno, DMc, DMc, DMc, ao_w, DMc, DMc, xa, xa, DMc, DMc, Tc, 0, 0);

  // 8) final norm, router, fused MoE (+residual) -> d_out
  k_rmsnorm<<<ROWSc, DMc, 0, stream>>>(xa, fn_w, xfn);
  k_router<<<ROWSc / 256, 256, 0, stream>>>(xfn, router_w, wfull);
  k_moe<<<ROWSc / 16, 256, 0, stream>>>(xfn, xa, wfull, exp_w1, exp_w2, outp);
}
